// TileGAT_47287589929667
// MI455X (gfx1250) — matrix operand order repaired and checked by reading, not compiled, not run on hardware
//
#include <hip/hip_runtime.h>
#include <hip/hip_bf16.h>
#include <math.h>

// Problem constants (match reference)
#define NN    50000
#define EE    800000
#define DIMM  256
#define NH    4
#define DH    64          // DIM / HEADS
#define RP    16
#define ASTR  (2*DH+RP)   // 144: per-head attention-vector stride

typedef __attribute__((ext_vector_type(16))) __bf16 v16bf;
typedef __attribute__((ext_vector_type(8)))  float  v8f;

// ---------------------------------------------------------------------------
// helpers
// ---------------------------------------------------------------------------
__device__ inline v16bf pack_bf16(float4 x0, float4 x1, float4 y0, float4 y1) {
  v16bf a;
  a[0]  = (__bf16)x0.x; a[1]  = (__bf16)x0.y; a[2]  = (__bf16)x0.z; a[3]  = (__bf16)x0.w;
  a[4]  = (__bf16)x1.x; a[5]  = (__bf16)x1.y; a[6]  = (__bf16)x1.z; a[7]  = (__bf16)x1.w;
  a[8]  = (__bf16)y0.x; a[9]  = (__bf16)y0.y; a[10] = (__bf16)y0.z; a[11] = (__bf16)y0.w;
  a[12] = (__bf16)y1.x; a[13] = (__bf16)y1.y; a[14] = (__bf16)y1.z; a[15] = (__bf16)y1.w;
  return a;
}

__global__ void zero_kernel(float* __restrict__ p, long n) {
  long i = (long)blockIdx.x * blockDim.x + threadIdx.x;
  long stride = (long)gridDim.x * blockDim.x;
  for (; i < n; i += stride) p[i] = 0.0f;
}

// ---------------------------------------------------------------------------
// Kernel 1: zh[h][n][k] = z[n][:] . W[h][:][k]     (bf16 WMMA, f32 accumulate)
// grid = (ceil(N/128), NH), block = 256 (8 wave32; each wave owns 16 rows x 64 cols)
// ---------------------------------------------------------------------------
__global__ __launch_bounds__(256)
void gemm_zh_kernel(const float* __restrict__ z, const float* __restrict__ W,
                    float* __restrict__ zh) {
  __shared__ __bf16 WT[DH][DIMM + 16];   // W[h]^T (n,k) bf16, row stride 32B-aligned
  const int h = blockIdx.y;
  const float* Wh = W + (size_t)h * DIMM * DH;
  for (int idx = threadIdx.x; idx < DIMM * DH; idx += 256) {
    int n = idx & (DH - 1);
    int k = idx >> 6;
    WT[n][k] = (__bf16)Wh[(size_t)k * DH + n];
  }
  __syncthreads();

  const int wave = threadIdx.x >> 5;
  const int lane = threadIdx.x & 31;
  const int m    = lane & 15;
  const int kh   = lane >> 4;                 // K-half held by this lane (contiguous K order)
  const int rowbase = blockIdx.x * 128 + wave * 16;
  const int row   = rowbase + m;
  const int row_c = row < NN ? row : NN - 1;
  const float* zr = z + (size_t)row_c * DIMM;

  v8f acc[4] = {};
#pragma unroll
  for (int kt = 0; kt < DIMM / 32; ++kt) {
    const int kb = kt * 32;
    if (kt + 2 < DIMM / 32) __builtin_prefetch(zr + kb + 64, 0, 3);
    // A fragment (16x32 bf16): lanes 0-15 hold K=0..15, lanes 16-31 K=16..31
    float4 x0 = *(const float4*)(zr + kb + kh * 16);
    float4 x1 = *(const float4*)(zr + kb + kh * 16 + 4);
    float4 y0 = *(const float4*)(zr + kb + kh * 16 + 8);
    float4 y1 = *(const float4*)(zr + kb + kh * 16 + 12);
    v16bf af = pack_bf16(x0, x1, y0, y1);
#pragma unroll
    for (int nt = 0; nt < 4; ++nt) {
      // B fragment (32x16 bf16): lane holds col n=m, 16 consecutive K from its K-half
      const v16bf bf = *(const v16bf*)&WT[nt * 16 + m][kb + kh * 16];
      acc[nt] = __builtin_amdgcn_wmma_f32_16x16x32_bf16(
          false, af, false, bf, (short)0, acc[nt], false, false);
    }
  }

  float* zhh = zh + (size_t)h * NN * DH;
#pragma unroll
  for (int nt = 0; nt < 4; ++nt)
#pragma unroll
    for (int r = 0; r < 8; ++r) {
      int ro = rowbase + kh * 8 + r;          // C layout: lanes 16-31 hold M=8..15
      if (ro < NN) zhh[(size_t)ro * DH + nt * 16 + m] = acc[nt][r];
    }
}

// ---------------------------------------------------------------------------
// Kernel 2: c_src[h,n] = zh[h,n,:].a_src[h], c_dst likewise
// ---------------------------------------------------------------------------
__global__ void csd_kernel(const float* __restrict__ zh, const float* __restrict__ a,
                           float* __restrict__ c_src, float* __restrict__ c_dst) {
  int idx = blockIdx.x * 256 + threadIdx.x;     // idx = h*N + n
  if (idx >= NH * NN) return;
  int h = idx / NN;
  const float* zr = zh + (size_t)idx * DH;
  const float* ah = a + h * ASTR;
  float s = 0.f, d = 0.f;
#pragma unroll 8
  for (int k = 0; k < DH; ++k) {
    float v = zr[k];
    s += v * ah[k];
    d += v * ah[DH + k];
  }
  c_src[idx] = s;
  c_dst[idx] = d;
}

// ---------------------------------------------------------------------------
// Kernel 3: per-edge rel-MLP, logits, exp, and denom scatter (softmax denominator)
// ---------------------------------------------------------------------------
__global__ void edge_kernel(const float* __restrict__ rel_off,
                            const float* __restrict__ rW1, const float* __restrict__ rb1,
                            const float* __restrict__ rW2, const float* __restrict__ rb2,
                            const float* __restrict__ a,
                            const int* __restrict__ e_src, const int* __restrict__ e_dst,
                            const float* __restrict__ c_src, const float* __restrict__ c_dst,
                            float* __restrict__ expe, float* __restrict__ denom) {
  int e = blockIdx.x * 256 + threadIdx.x;
  if (e >= EE) return;
  float ox = rel_off[2 * e], oy = rel_off[2 * e + 1];
  float hid[RP], rel[RP];
#pragma unroll
  for (int p = 0; p < RP; ++p) {
    float v = ox * rW1[p] + oy * rW1[RP + p] + rb1[p];
    hid[p] = v > 0.f ? v : 0.f;
  }
#pragma unroll
  for (int p = 0; p < RP; ++p) {
    float v = rb2[p];
#pragma unroll
    for (int q = 0; q < RP; ++q) v += hid[q] * rW2[q * RP + p];
    rel[p] = v;
  }
  int s = e_src[e], d = e_dst[e];
#pragma unroll
  for (int h = 0; h < NH; ++h) {
    const float* ar = a + h * ASTR + 2 * DH;
    float cr = 0.f;
#pragma unroll
    for (int p = 0; p < RP; ++p) cr += rel[p] * ar[p];
    float x = c_src[h * NN + s] + c_dst[h * NN + d] + cr;
    x = x > 0.f ? x : 0.2f * x;               // leaky_relu(0.2)
    float ex = __expf(x);
    expe[(size_t)h * EE + e] = ex;
    atomicAdd(&denom[h * NN + d], ex);
  }
}

// ---------------------------------------------------------------------------
// Kernel 4: agg[h][dst][:] += alpha * zh[h][src][:]   (one wave32 per (edge,head))
// ---------------------------------------------------------------------------
__global__ __launch_bounds__(256)
void agg_kernel(const float* __restrict__ zh, const float* __restrict__ expe,
                const float* __restrict__ denom,
                const int* __restrict__ e_src, const int* __restrict__ e_dst,
                float* __restrict__ agg) {
  long wid = (long)blockIdx.x * 8 + (threadIdx.x >> 5);
  if (wid >= (long)EE * NH) return;
  int e = (int)(wid >> 2);
  int h = (int)(wid & 3);
  int lane = threadIdx.x & 31;
  int s = e_src[e], d = e_dst[e];
  float alpha = expe[(size_t)h * EE + e] / (denom[h * NN + d] + 1e-6f);
  const float* zs = zh + ((size_t)h * NN + s) * DH;
  float* ad = agg + ((size_t)h * NN + d) * DH;
  atomicAdd(&ad[lane],      zs[lane]      * alpha);
  atomicAdd(&ad[lane + 32], zs[lane + 32] * alpha);
}

// ---------------------------------------------------------------------------
// Kernel 5: out = concat_heads(agg) @ proj_W + proj_b + z   (bf16 WMMA)
// grid = (ceil(N/128), 4 column quarters), block = 256
// ---------------------------------------------------------------------------
__global__ __launch_bounds__(256)
void gemm_out_kernel(const float* __restrict__ agg, const float* __restrict__ pW,
                     const float* __restrict__ pb, const float* __restrict__ z,
                     float* __restrict__ out) {
  __shared__ __bf16 PT[DH][DIMM + 16];  // proj_W[:, colbase:colbase+64]^T as bf16
  const int colbase = blockIdx.y * 64;
  for (int idx = threadIdx.x; idx < DIMM * DH; idx += 256) {
    int c = idx & (DH - 1);
    int k = idx >> 6;
    PT[c][k] = (__bf16)pW[(size_t)k * DIMM + colbase + c];
  }
  __syncthreads();

  const int wave = threadIdx.x >> 5;
  const int lane = threadIdx.x & 31;
  const int m    = lane & 15;
  const int kh   = lane >> 4;
  const int rowbase = blockIdx.x * 128 + wave * 16;
  const int row   = rowbase + m;
  const int row_c = row < NN ? row : NN - 1;

  v8f acc[4] = {};
#pragma unroll
  for (int kt = 0; kt < DIMM / 32; ++kt) {
    const int kb = kt * 32;
    const int h  = kb >> 6;                    // head providing this K-slab
    const int ko = kb & (DH - 1);
    const float* ar = agg + ((size_t)h * NN + row_c) * DH + ko;
    if (kt + 2 < DIMM / 32) __builtin_prefetch(ar + 64, 0, 3);
    float4 x0 = *(const float4*)(ar + kh * 16);
    float4 x1 = *(const float4*)(ar + kh * 16 + 4);
    float4 y0 = *(const float4*)(ar + kh * 16 + 8);
    float4 y1 = *(const float4*)(ar + kh * 16 + 12);
    v16bf af = pack_bf16(x0, x1, y0, y1);
#pragma unroll
    for (int nt = 0; nt < 4; ++nt) {
      const v16bf bf = *(const v16bf*)&PT[nt * 16 + m][kb + kh * 16];
      acc[nt] = __builtin_amdgcn_wmma_f32_16x16x32_bf16(
          false, af, false, bf, (short)0, acc[nt], false, false);
    }
  }

#pragma unroll
  for (int nt = 0; nt < 4; ++nt)
#pragma unroll
    for (int r = 0; r < 8; ++r) {
      int ro  = rowbase + kh * 8 + r;
      int col = colbase + nt * 16 + m;
      if (ro < NN)
        out[(size_t)ro * DIMM + col] = acc[nt][r] + pb[col] + z[(size_t)ro * DIMM + col];
    }
}

// ---------------------------------------------------------------------------
// launch
// ---------------------------------------------------------------------------
extern "C" void kernel_launch(void* const* d_in, const int* in_sizes, int n_in,
                              void* d_out, int out_size, void* d_ws, size_t ws_size,
                              hipStream_t stream) {
  const float* z       = (const float*)d_in[0];
  const float* rel_off = (const float*)d_in[1];
  const float* W       = (const float*)d_in[2];
  const float* a       = (const float*)d_in[3];
  const float* rW1     = (const float*)d_in[4];
  const float* rb1     = (const float*)d_in[5];
  const float* rW2     = (const float*)d_in[6];
  const float* rb2     = (const float*)d_in[7];
  const float* pW      = (const float*)d_in[8];
  const float* pb      = (const float*)d_in[9];
  const int*   e_src   = (const int*)d_in[10];
  const int*   e_dst   = (const int*)d_in[11];
  float* out = (float*)d_out;

  // workspace carve-up (floats)
  float* ws   = (float*)d_ws;
  size_t o = 0;
  float* zh    = ws + o;  o += (size_t)NH * NN * DH;  // 51.2 MB
  float* csrc  = ws + o;  o += (size_t)NH * NN;
  float* cdst  = ws + o;  o += (size_t)NH * NN;
  float* denom = ws + o;  o += (size_t)NH * NN;       // atomic target (zeroed)
  float* expe  = ws + o;  o += (size_t)NH * EE;
  float* agg   = ws + o;  o += (size_t)NH * NN * DH;  // atomic target (zeroed)

  // zero the atomic-accumulation region: [denom .. agg end) is contiguous
  long nz = (long)NH * NN + (long)NH * EE + (long)NH * NN * DH;
  zero_kernel<<<2048, 256, 0, stream>>>(denom, nz);

  dim3 g1((NN + 127) / 128, NH);
  gemm_zh_kernel<<<g1, 256, 0, stream>>>(z, W, zh);

  csd_kernel<<<(NH * NN + 255) / 256, 256, 0, stream>>>(zh, a, csrc, cdst);

  edge_kernel<<<(EE + 255) / 256, 256, 0, stream>>>(rel_off, rW1, rb1, rW2, rb2, a,
                                                    e_src, e_dst, csrc, cdst,
                                                    expe, denom);

  long waves = (long)EE * NH;
  agg_kernel<<<(int)((waves + 7) / 8), 256, 0, stream>>>(zh, expe, denom,
                                                         e_src, e_dst, agg);

  dim3 g5((NN + 127) / 128, DIMM / 64);
  gemm_out_kernel<<<g5, 256, 0, stream>>>(agg, pW, pb, z, out);
}
